// VectorQuantizer_61787399520295
// MI455X (gfx1250) — matrix operand order repaired and checked by reading, not compiled, not run on hardware
//
#include <hip/hip_runtime.h>
#include <hip/hip_bf16.h>

typedef __attribute__((ext_vector_type(16))) __bf16 v16bf;
typedef __attribute__((ext_vector_type(8)))  __bf16 v8bf;
typedef __attribute__((ext_vector_type(8)))  float  v8f;
typedef __attribute__((ext_vector_type(4)))  unsigned int uint4v;
typedef __attribute__((ext_vector_type(8)))  int    int8v;
typedef __attribute__((ext_vector_type(4)))  int    int4v;

#define NROWS 65536
#define DDIM  256
#define KDICT 1024
#define NELEM (NROWS * DDIM)   // 16777216

#if defined(__has_builtin)
#  if __has_builtin(__builtin_amdgcn_tensor_load_to_lds) && \
      __has_builtin(__builtin_amdgcn_s_wait_tensorcnt)
#    define VQ_USE_TDM 1
#  endif
#endif
#ifndef VQ_USE_TDM
#  define VQ_USE_TDM 0
#endif

// ---------- helpers ----------
__device__ __forceinline__ unsigned short f2bf_bits(float f) {
    unsigned int u = __float_as_uint(f);
    unsigned int r = u + 0x7FFFu + ((u >> 16) & 1u);   // round-to-nearest-even
    return (unsigned short)(r >> 16);
}

// ---------- Pass A: codebook -> bf16 copy + ||c||^2 ----------
__global__ void vq_cbprep_kernel(const float* __restrict__ cb,
                                 unsigned short* __restrict__ cb_bf,
                                 float* __restrict__ cnorm) {
    __shared__ float red[8];
    const int row = blockIdx.x;           // one block per codebook row
    const int t   = threadIdx.x;          // 256 threads
    float v = cb[row * DDIM + t];
    cb_bf[row * DDIM + t] = f2bf_bits(v);
    float s = v * v;
    #pragma unroll
    for (int m = 16; m >= 1; m >>= 1) s += __shfl_xor(s, m, 32);
    const int lane = t & 31, wave = t >> 5;
    if (lane == 0) red[wave] = s;
    __syncthreads();
    if (t == 0) {
        float acc = 0.f;
        #pragma unroll
        for (int w = 0; w < 8; ++w) acc += red[w];
        cnorm[row] = acc;
    }
}

// ---------- Pass B: WMMA distance + argmin ----------
// 8 waves/block, 16 rows/wave -> 128 rows/block, 512 blocks.
// LDS tile: 64 codebook rows x 256 bf16, padded stride 264 bf16 per row
// (TDM pad: 4 DWORDs after every 128 DWORDs -> identical layout in HW).
#define CB_STRIDE 264
#define TILE_USHORTS (64 * CB_STRIDE)

#if VQ_USE_TDM
// Issue one TDM load of a 64x256 bf16 tile (row stride 256) into LDS with
// padding. D# per CDNA5 ISA ch.8.3/8.4. Uniform operands; EXEC ignored.
__device__ __forceinline__ void tdm_load_tile(const unsigned short* gsrc,
                                              unsigned int ldsByte) {
    unsigned long long ga = (unsigned long long)(const void*)gsrc;
    uint4v g0;
    g0[0] = 1u;                                          // count=1, user D#
    g0[1] = ldsByte;                                     // lds_addr
    g0[2] = (unsigned int)(ga & 0xFFFFFFFFull);          // global_addr[31:0]
    g0[3] = (unsigned int)((ga >> 32) & 0x1FFFFFFull)    // global_addr[56:32]
            | 0x80000000u;                               // type=2 ("image")
    int8v g1;
    g1[0] = (int)((1u << 16)      // data_size = 2 bytes
            | (1u << 20)          // pad_enable
            | (6u << 22)          // pad_interval: 128 DWORDs (=512B row)
            | (3u << 25));        // pad_amount: 4 DWORDs (=8 bf16) -> stride 264
    g1[1] = (int)(256u << 16);    // tensor_dim0 = 256
    g1[2] = (int)(1024u << 16);   // tensor_dim1 = 1024
    g1[3] = (int)(256u << 16);    // tile_dim0 = 256
    g1[4] = 64;                   // tile_dim1 = 64, tile_dim2 = 0
    g1[5] = 256;                  // tensor_dim0_stride = 256
    g1[6] = 0;
    g1[7] = 0;
    int4v gz4 = {0, 0, 0, 0};             // groups 2/3 unused (2D tensor)
    int8v gz8 = {0, 0, 0, 0, 0, 0, 0, 0}; // extra 256-bit operand (6-arg form)
    __builtin_amdgcn_tensor_load_to_lds(g0, g1, gz4, gz4, gz8, 0);
}
#define NBUF 2
#else
#define NBUF 1
#endif

__global__ void __launch_bounds__(256)
vq_argmin_kernel(const float* __restrict__ z,
                 const unsigned short* __restrict__ cb_bf,
                 const float* __restrict__ cnorm,
                 int* __restrict__ idxOut) {
    __shared__ __align__(16) unsigned short ldsB[NBUF][TILE_USHORTS];

    const int tid  = threadIdx.x;
    const int lane = tid & 31;
    const int wave = tid >> 5;
    const int rowBase = blockIdx.x * 128 + wave * 16;
    const int lrow = lane & 15;              // row / column slot within 16
    const int hi   = (lane >> 4) & 1;        // lane half selector

    // --- load the wave's 16 z rows once, pack A fragments (16x32 bf16 ISA layout)
    v16bf afrag[8];
    {
        const float* zr = z + (rowBase + lrow) * DDIM;
        #pragma unroll
        for (int kc = 0; kc < 8; ++kc) {
            const int koff = kc * 32 + (hi ? 8 : 0);
            float4 a0 = *(const float4*)(zr + koff);
            float4 a1 = *(const float4*)(zr + koff + 4);
            float4 a2 = *(const float4*)(zr + koff + 16);
            float4 a3 = *(const float4*)(zr + koff + 20);
            union { v16bf v; unsigned short u[16]; } pk;
            pk.u[0]=f2bf_bits(a0.x); pk.u[1]=f2bf_bits(a0.y);
            pk.u[2]=f2bf_bits(a0.z); pk.u[3]=f2bf_bits(a0.w);
            pk.u[4]=f2bf_bits(a1.x); pk.u[5]=f2bf_bits(a1.y);
            pk.u[6]=f2bf_bits(a1.z); pk.u[7]=f2bf_bits(a1.w);
            pk.u[8]=f2bf_bits(a2.x); pk.u[9]=f2bf_bits(a2.y);
            pk.u[10]=f2bf_bits(a2.z); pk.u[11]=f2bf_bits(a2.w);
            pk.u[12]=f2bf_bits(a3.x); pk.u[13]=f2bf_bits(a3.y);
            pk.u[14]=f2bf_bits(a3.z); pk.u[15]=f2bf_bits(a3.w);
            afrag[kc] = pk.v;
        }
    }

    float bestd[8];
    int   besti[8];
    #pragma unroll
    for (int r = 0; r < 8; ++r) { bestd[r] = 3.4e38f; besti[r] = 0; }

#if VQ_USE_TDM
    if (wave == 0) {                         // prologue: tile 0 in flight
        tdm_load_tile(cb_bf,
                      (unsigned int)(unsigned long long)(const void*)&ldsB[0][0]);
    }
#else
    const unsigned int* cbU = (const unsigned int*)cb_bf;  // 128 uints per row
#endif

    for (int mt = 0; mt < 16; ++mt) {        // 16 macro-tiles of 64 codebook rows
#if VQ_USE_TDM
        if (wave == 0) __builtin_amdgcn_s_wait_tensorcnt(0);
        __syncthreads();                     // tile mt visible to all waves
        if (wave == 0 && mt < 15) {          // overlap DMA of mt+1 with compute
            tdm_load_tile(cb_bf + (mt + 1) * 64 * DDIM,
                (unsigned int)(unsigned long long)(const void*)&ldsB[(mt + 1) & 1][0]);
        }
        const unsigned short* tile = &ldsB[mt & 1][0];
#else
        __syncthreads();
        unsigned int* ldsU = (unsigned int*)&ldsB[0][0];
        #pragma unroll 4
        for (int i = tid; i < 64 * 128; i += 256) {
            const int col = i >> 7;
            const int q   = i & 127;
            ldsU[col * (CB_STRIDE / 2) + q] = cbU[(mt * 64 + col) * (DDIM / 2) + q];
        }
        __syncthreads();
        const unsigned short* tile = &ldsB[0][0];
#endif

        #pragma unroll
        for (int s = 0; s < 4; ++s) {
            const int n0 = mt * 64 + s * 16;
            const float cn = cnorm[n0 + lrow];        // L2-resident scalar
            const int colBase = (s * 16 + lrow) * CB_STRIDE + (hi ? 8 : 0);

            // 2-deep software pipeline: each B fragment's ds loads are issued
            // two WMMAs ahead of use, so waits become partial (not dscnt==0).
            #define LOADFRAG(kc) ({                                           \
                const v8bf* bp_ = (const v8bf*)(&tile[colBase + (kc) * 32]);  \
                v8bf lo_ = bp_[0];                                            \
                v8bf hi_ = bp_[2];                                            \
                __builtin_shufflevector(lo_, hi_,                             \
                    0,1,2,3,4,5,6,7,8,9,10,11,12,13,14,15); })

            v16bf bcur = LOADFRAG(0);
            v16bf bnxt = LOADFRAG(1);
            v8f acc0 = {};
            v8f acc1 = {};
            #pragma unroll
            for (int kc = 0; kc < 8; ++kc) {
                v16bf bpre = bcur;                     // placeholder
                if (kc < 6) bpre = LOADFRAG(kc + 2);
                if (kc & 1) {
                    acc1 = __builtin_amdgcn_wmma_f32_16x16x32_bf16(
                            false, afrag[kc], false, bcur, (short)0, acc1,
                            false, false);
                } else {
                    acc0 = __builtin_amdgcn_wmma_f32_16x16x32_bf16(
                            false, afrag[kc], false, bcur, (short)0, acc0,
                            false, false);
                }
                bcur = bnxt;
                bnxt = bpre;
            }
            #undef LOADFRAG

            const int n = n0 + lrow;
            #pragma unroll
            for (int r = 0; r < 8; ++r) {
                float d = fmaf(-2.0f, acc0[r] + acc1[r], cn);  // ||c||^2 - 2 z.c
                if (d < bestd[r]) { bestd[r] = d; besti[r] = n; }
            }
        }
        __syncthreads();                     // tile fully consumed before reuse
    }

    // reduce argmin across the 16 column-lanes (halves stay separate)
    #pragma unroll
    for (int m = 8; m >= 1; m >>= 1) {
        #pragma unroll
        for (int r = 0; r < 8; ++r) {
            float od = __shfl_xor(bestd[r], m, 32);
            int   oi = __shfl_xor(besti[r], m, 32);
            if (od < bestd[r] || (od == bestd[r] && oi < besti[r])) {
                bestd[r] = od; besti[r] = oi;
            }
        }
    }
    if (lrow == 0) {   // lane 0 -> rows M=0..7, lane 16 -> rows M=8..15
        const int mbase = rowBase + (hi ? 8 : 0);
        #pragma unroll
        for (int r = 0; r < 8; ++r) idxOut[mbase + r] = besti[r];
    }
}

// ---------- Pass Z: zero accumulators ----------
__global__ void vq_zero_kernel(float* __restrict__ counts, float* __restrict__ lossAcc) {
    const int t = blockIdx.x * blockDim.x + threadIdx.x;
    if (t < KDICT) counts[t] = 0.0f;
    if (t == 0) lossAcc[0] = 0.0f;
}

// ---------- Pass C: gather z_q, (z_q - z)^2 partial sum, histogram ----------
__global__ void __launch_bounds__(256)
vq_gather_kernel(const float* __restrict__ z,
                 const float* __restrict__ cb,
                 const int* __restrict__ idxArr,
                 float* __restrict__ zq,
                 float* __restrict__ counts,
                 float* __restrict__ lossAcc) {
    __shared__ float red[8];
    const int lane = threadIdx.x & 31;
    const int wave = threadIdx.x >> 5;
    const int row  = blockIdx.x * 8 + wave;      // one wave per row
    const int ci   = idxArr[row];

    const float* zr = z  + row * DDIM + lane * 8;
    const float* cr = cb + ci  * DDIM + lane * 8;
    float*       qr = zq + row * DDIM + lane * 8;

    float4 c0 = *(const float4*)(cr);
    float4 c1 = *(const float4*)(cr + 4);
    float4 z0 = *(const float4*)(zr);
    float4 z1 = *(const float4*)(zr + 4);
    *(float4*)(qr)     = c0;
    *(float4*)(qr + 4) = c1;

    float s = 0.f;
    float d;
    d = c0.x - z0.x; s = fmaf(d, d, s);
    d = c0.y - z0.y; s = fmaf(d, d, s);
    d = c0.z - z0.z; s = fmaf(d, d, s);
    d = c0.w - z0.w; s = fmaf(d, d, s);
    d = c1.x - z1.x; s = fmaf(d, d, s);
    d = c1.y - z1.y; s = fmaf(d, d, s);
    d = c1.z - z1.z; s = fmaf(d, d, s);
    d = c1.w - z1.w; s = fmaf(d, d, s);

    #pragma unroll
    for (int m = 16; m >= 1; m >>= 1) s += __shfl_xor(s, m, 32);
    if (lane == 0) {
        red[wave] = s;
        atomicAdd(&counts[ci], 1.0f);
    }
    __syncthreads();
    if (threadIdx.x == 0) {
        float t = 0.f;
        #pragma unroll
        for (int w = 0; w < 8; ++w) t += red[w];
        atomicAdd(lossAcc, t);
    }
}

// ---------- Pass D: loss + perplexity scalars ----------
__global__ void vq_finalize_kernel(const float* __restrict__ counts,
                                   const float* __restrict__ lossAcc,
                                   float* __restrict__ out) {
    __shared__ float red[32];
    const int t = threadIdx.x;               // 1024 threads
    const float p = counts[t] * (1.0f / (float)NROWS);
    float e = p * logf(p + 1e-10f);
    #pragma unroll
    for (int m = 16; m >= 1; m >>= 1) e += __shfl_xor(e, m, 32);
    const int lane = t & 31, wave = t >> 5;
    if (lane == 0) red[wave] = e;
    __syncthreads();
    if (wave == 0) {
        float v = red[lane];
        #pragma unroll
        for (int m = 16; m >= 1; m >>= 1) v += __shfl_xor(v, m, 32);
        if (lane == 0) {
            out[NELEM]     = lossAcc[0] * (1.0f + 0.001f) / (float)NELEM; // loss
            out[NELEM + 1] = expf(-v);                                    // perplexity
        }
    }
}

extern "C" void kernel_launch(void* const* d_in, const int* in_sizes, int n_in,
                              void* d_out, int out_size, void* d_ws, size_t ws_size,
                              hipStream_t stream) {
    const float* z  = (const float*)d_in[0];   // [65536, 256]
    const float* cb = (const float*)d_in[1];   // [1024, 256]
    float* out = (float*)d_out;                // z_q (16777216) + loss + perplexity

    // workspace layout
    char* ws = (char*)d_ws;
    unsigned short* cb_bf = (unsigned short*)(ws);                 // 512 KB
    float* cnorm   = (float*)(ws + 524288);                        // 4 KB
    int*   idxArr  = (int*)  (ws + 524288 + 4096);                 // 256 KB
    float* counts  = (float*)(ws + 524288 + 4096 + 262144);        // 4 KB
    float* lossAcc = (float*)(ws + 524288 + 4096 + 262144 + 4096); // 4 B

    vq_zero_kernel<<<1, 1024, 0, stream>>>(counts, lossAcc);
    vq_cbprep_kernel<<<KDICT, 256, 0, stream>>>(cb, cb_bf, cnorm);
    vq_argmin_kernel<<<NROWS / 128, 256, 0, stream>>>(z, cb_bf, cnorm, idxArr);
    vq_gather_kernel<<<NROWS / 8, 256, 0, stream>>>(z, cb, idxArr, out,
                                                    counts, lossAcc);
    vq_finalize_kernel<<<1, 1024, 0, stream>>>(counts, lossAcc, out);
}
